// AttentionFlowLayer_23252952940902
// MI455X (gfx1250) — compile-verified, matrix operand order checked by reading, not run on hardware
//
#include <hip/hip_runtime.h>
#include <hip/hip_bf16.h>

typedef __attribute__((ext_vector_type(16))) _Float16 v16h;
typedef __attribute__((ext_vector_type(8)))  _Float16 v8h;
typedef __attribute__((ext_vector_type(8)))  float    v8f;

#define TT   1024
#define JJ   128
#define BB   32
#define DD   256
#define TBLK 128
#define VERY_NEG (-1.0e30f)

// LDS layout (bytes):
//   aw   : 0       .. 65536   (128 x 256 f16)  ctx*w_cq tile; reused for P (16x128 per wave)
//   qh   : 65536   .. 131072  (128 x 256 f16)  question_b
//   qT   : 131072  .. 196608  (256 x 128 f16)  question_b transposed
//   cproj: 196608  .. 197120  (128 f32)
//   qproj: 197120  .. 197632  (128 f32)
#define SMEM_BYTES 197632

__device__ __forceinline__ float wave_sum(float v) {
    v += __shfl_xor(v, 16, 32); v += __shfl_xor(v, 8, 32);
    v += __shfl_xor(v, 4, 32);  v += __shfl_xor(v, 2, 32);
    v += __shfl_xor(v, 1, 32);
    return v;
}
__device__ __forceinline__ float half_max(float v) {   // reduce within 16-lane half
    v = fmaxf(v, __shfl_xor(v, 8, 32)); v = fmaxf(v, __shfl_xor(v, 4, 32));
    v = fmaxf(v, __shfl_xor(v, 2, 32)); v = fmaxf(v, __shfl_xor(v, 1, 32));
    return v;
}
__device__ __forceinline__ float half_sum(float v) {
    v += __shfl_xor(v, 8, 32); v += __shfl_xor(v, 4, 32);
    v += __shfl_xor(v, 2, 32); v += __shfl_xor(v, 1, 32);
    return v;
}

// Load a v16h A/B fragment for wmma f32_16x16x32_f16 from an LDS row (row-major,
// contiguous in K). hs = lane>>4 selects the K sub-chunk per ISA 7.12.2:
//   lanes 0-15 : K = k0+0..7  and k0+16..23
//   lanes16-31 : K = k0+8..15 and k0+24..31
__device__ __forceinline__ v16h load_frag(const _Float16* rowk0, int hs) {
    v8h lo = *(const v8h*)(rowk0 + hs * 8);
    v8h hi = *(const v8h*)(rowk0 + 16 + hs * 8);
    return __builtin_shufflevector(lo, hi, 0,1,2,3,4,5,6,7,8,9,10,11,12,13,14,15);
}

__global__ void __launch_bounds__(256)
bidaf_main(const float* __restrict__ ctx, const float* __restrict__ qst,
           const int* __restrict__ clen, const int* __restrict__ qlen,
           const float* __restrict__ attw, float* __restrict__ out,
           float* __restrict__ mS) {
    extern __shared__ char smem[];
    _Float16* aw  = (_Float16*)smem;
    _Float16* qh  = (_Float16*)(smem + 65536);
    _Float16* qT  = (_Float16*)(smem + 131072);
    float* cproj  = (float*)(smem + 196608);
    float* qproj  = (float*)(smem + 197120);

    const int b   = blockIdx.x;
    const int tb  = blockIdx.y;          // t-block index (128 rows each)
    const int tid = threadIdx.x;
    const int wv  = tid >> 5;
    const int ln  = tid & 31;
    const int d0  = ln * 8;

    // ---------------- Stage question rows (16 per wave) ----------------
    for (int rr = 0; rr < 16; ++rr) {
        const int j = wv * 16 + rr;
        const float* qb = qst + ((size_t)j * BB + b) * DD + d0;
        float f[8], wq[8];
        *(float4*)(f)     = *(const float4*)(qb);
        *(float4*)(f + 4) = *(const float4*)(qb + 4);
        *(float4*)(wq)     = *(const float4*)(attw + DD + d0);
        *(float4*)(wq + 4) = *(const float4*)(attw + DD + d0 + 4);
        float qp = 0.f;
        v8h hv;
        #pragma unroll
        for (int i = 0; i < 8; ++i) {
            qp += f[i] * wq[i];
            hv[i] = (_Float16)f[i];
        }
        qp = wave_sum(qp);
        if (ln == 0) qproj[j] = qp;
        *(v8h*)(qh + j * 256 + d0) = hv;
        #pragma unroll
        for (int i = 0; i < 8; ++i) qT[(d0 + i) * 128 + j] = hv[i];
    }

    // ---------------- Stage context rows (16 per wave): aw = ctx*w_cq ----------------
    for (int rr = 0; rr < 16; ++rr) {
        const int trl = wv * 16 + rr;
        const int tg  = tb * TBLK + trl;
        const float* cb = ctx + ((size_t)tg * BB + b) * DD + d0;
        float f[8], wc[8], wcq[8];
        *(float4*)(f)      = *(const float4*)(cb);
        *(float4*)(f + 4)  = *(const float4*)(cb + 4);
        *(float4*)(wc)     = *(const float4*)(attw + d0);
        *(float4*)(wc + 4) = *(const float4*)(attw + d0 + 4);
        *(float4*)(wcq)     = *(const float4*)(attw + 2 * DD + d0);
        *(float4*)(wcq + 4) = *(const float4*)(attw + 2 * DD + d0 + 4);
        float cp = 0.f;
        v8h hv;
        #pragma unroll
        for (int i = 0; i < 8; ++i) {
            cp += f[i] * wc[i];
            hv[i] = (_Float16)(f[i] * wcq[i]);
        }
        cp = wave_sum(cp);
        if (ln == 0) cproj[trl] = cp;
        *(v8h*)(aw + trl * 256 + d0) = hv;
    }

    __syncthreads();

    const int hs = ln >> 4;     // lane half
    const int mr = ln & 15;     // A-row / B-column within tile
    const int cl = clen[b];
    const int ql = qlen[b];

    // ---------------- S = A(16x256) * Q^T  via 8 j-tiles, K-steps of 32 ----------------
    v8f accS[8];
    #pragma unroll
    for (int jt = 0; jt < 8; ++jt) accS[jt] = (v8f){0,0,0,0,0,0,0,0};

    const _Float16* arow = aw + (wv * 16 + mr) * 256;
    #pragma unroll
    for (int kk = 0; kk < 8; ++kk) {
        const int k0 = kk * 32;
        v16h av = load_frag(arow + k0, hs);
        #pragma unroll
        for (int jt = 0; jt < 8; ++jt) {
            const _Float16* brow = qh + (jt * 16 + mr) * 256;
            v16h bv = load_frag(brow + k0, hs);
            accS[jt] = __builtin_amdgcn_wmma_f32_16x16x32_f16(
                false, av, false, bv, (short)0, accS[jt], false, false);
        }
    }

    // ---------------- bias + mask + softmax over j (per row), record rowmax ----------------
    #pragma unroll
    for (int r = 0; r < 8; ++r) {
        const int trl = wv * 16 + r + 8 * hs;      // C-layout: M = r + 8*(lane>=16)
        const int tg  = tb * TBLK + trl;
        const float cp = cproj[trl];
        float rmax = -3.0e38f;
        #pragma unroll
        for (int jt = 0; jt < 8; ++jt) {
            const int j = jt * 16 + mr;            // N = lane & 15
            float s = accS[jt][r] + cp + qproj[j];
            if (!((tg < cl) && (j < ql))) s += VERY_NEG;
            accS[jt][r] = s;
            rmax = fmaxf(rmax, s);
        }
        rmax = half_max(rmax);
        if (mr == 0) mS[(size_t)b * TT + tg] = rmax;
        float es = 0.f;
        #pragma unroll
        for (int jt = 0; jt < 8; ++jt) {
            float e = __expf(accS[jt][r] - rmax);
            accS[jt][r] = e;
            es += e;
        }
        es = half_sum(es);
        const float inv = 1.0f / es;
        #pragma unroll
        for (int jt = 0; jt < 8; ++jt) accS[jt][r] *= inv;
    }

    // ---------------- store P (f16) into this wave's private aw rows ----------------
    #pragma unroll
    for (int r = 0; r < 8; ++r) {
        const int trl = wv * 16 + r + 8 * hs;
        #pragma unroll
        for (int jt = 0; jt < 8; ++jt)
            aw[trl * 256 + jt * 16 + mr] = (_Float16)accS[jt][r];
    }

    // ---------------- c2q = P(16x128) * Q(128x256) ; write G0..G2 ----------------
    const _Float16* prow = aw + (wv * 16 + mr) * 256;
    for (int h = 0; h < 2; ++h) {
        v8f accC[8];
        #pragma unroll
        for (int n = 0; n < 8; ++n) accC[n] = (v8f){0,0,0,0,0,0,0,0};
        #pragma unroll
        for (int kc = 0; kc < 4; ++kc) {
            const int k0 = kc * 32;
            v16h av = load_frag(prow + k0, hs);
            #pragma unroll
            for (int n = 0; n < 8; ++n) {
                const _Float16* brow = qT + ((h * 8 + n) * 16 + mr) * 128;
                v16h bv = load_frag(brow + k0, hs);
                accC[n] = __builtin_amdgcn_wmma_f32_16x16x32_f16(
                    false, av, false, bv, (short)0, accC[n], false, false);
            }
        }
        #pragma unroll
        for (int n = 0; n < 8; ++n) {
            const int d = (h * 8 + n) * 16 + mr;
            #pragma unroll
            for (int r = 0; r < 8; ++r) {
                const int trl = wv * 16 + r + 8 * hs;
                const int tg  = tb * TBLK + trl;
                const float tm  = (tg < cl) ? 1.0f : 0.0f;
                const float cq  = accC[n][r];
                const float cv  = ctx[((size_t)tg * BB + b) * DD + d];
                const size_t ob = ((size_t)b * TT + tg) * (4 * DD);
                out[ob + d]            = cv * tm;
                out[ob + DD + d]       = cq * tm;
                out[ob + 2 * DD + d]   = cv * cq * tm;
            }
        }
    }
}

// ---------------- softmax over t of rowmax; q2c[b,d] = sum_t value * ctx ----------------
__global__ void __launch_bounds__(256)
bidaf_q2c(const float* __restrict__ ctx, const float* __restrict__ mS,
          float* __restrict__ q2c) {
    const int b = blockIdx.x, tid = threadIdx.x;
    __shared__ float sval[TT];
    __shared__ float sred[256];

    float lm = -3.0e38f;
    for (int t = tid; t < TT; t += 256) {
        float v = mS[(size_t)b * TT + t];
        sval[t] = v;
        lm = fmaxf(lm, v);
    }
    sred[tid] = lm; __syncthreads();
    for (int s = 128; s > 0; s >>= 1) {
        if (tid < s) sred[tid] = fmaxf(sred[tid], sred[tid + s]);
        __syncthreads();
    }
    const float gmax = sred[0]; __syncthreads();

    float ls = 0.f;
    for (int t = tid; t < TT; t += 256) {
        float e = __expf(sval[t] - gmax);
        sval[t] = e;
        ls += e;
    }
    sred[tid] = ls; __syncthreads();
    for (int s = 128; s > 0; s >>= 1) {
        if (tid < s) sred[tid] += sred[tid + s];
        __syncthreads();
    }
    const float inv = 1.0f / sred[0];

    const int d = tid;   // D == blockDim
    float acc = 0.f;
    for (int t = 0; t < TT; ++t)
        acc += sval[t] * ctx[((size_t)t * BB + b) * DD + d];
    q2c[b * DD + d] = acc * inv;
}

// ---------------- G3 = ctx * q2c (broadcast over t) * mask ----------------
__global__ void __launch_bounds__(256)
bidaf_g3(const float* __restrict__ ctx, const int* __restrict__ clen,
         const float* __restrict__ q2c, float* __restrict__ out) {
    const int t = blockIdx.x, b = blockIdx.y, d = threadIdx.x;
    const float tm = (t < clen[b]) ? 1.0f : 0.0f;
    const float v = ctx[((size_t)t * BB + b) * DD + d] * q2c[b * DD + d] * tm;
    out[((size_t)b * TT + t) * (4 * DD) + 3 * DD + d] = v;
}

extern "C" void kernel_launch(void* const* d_in, const int* in_sizes, int n_in,
                              void* d_out, int out_size, void* d_ws, size_t ws_size,
                              hipStream_t stream) {
    const float* ctx  = (const float*)d_in[0];   // (T,B,D)
    const float* qst  = (const float*)d_in[1];   // (J,B,D)
    const int*   clen = (const int*)d_in[2];     // (B,)
    const int*   qlen = (const int*)d_in[3];     // (B,)
    const float* attw = (const float*)d_in[4];   // (3D,1)
    float* out = (float*)d_out;                  // (B,T,4D)

    float* mS  = (float*)d_ws;                   // B*T floats
    float* q2c = mS + BB * TT;                   // B*D floats

    bidaf_main<<<dim3(BB, TT / TBLK), 256, SMEM_BYTES, stream>>>(
        ctx, qst, clen, qlen, attw, out, mS);
    bidaf_q2c<<<BB, 256, 0, stream>>>(ctx, mS, q2c);
    bidaf_g3<<<dim3(TT, BB), 256, 0, stream>>>(ctx, clen, q2c, out);
}
